// HybridSelectorModel_68418829025685
// MI455X (gfx1250) — matrix operand order repaired and checked by reading, not compiled, not run on hardware
//
#include <hip/hip_runtime.h>
#include <hip/hip_bf16.h>

typedef __attribute__((ext_vector_type(16))) _Float16 v16h;
typedef __attribute__((ext_vector_type(8)))  float    v8f;

#define B_TOT 32768
#define T_LEN 64
#define NN    4      // nodes
#define HID   64     // GRU hidden
#define MM    4      // modes / experts
#define HB    256    // bank hidden
#define BT    32     // batch rows per block (two 16-row WMMA tiles)
#define HSTR  104    // padded f16 row stride: [h(64)|x(4)|1|zeros], 16B aligned

// Native transcendentals (TRANS32 ops, co-execute with WMMA).
__device__ __forceinline__ float fast_rcp(float x) {
#if __has_builtin(__builtin_amdgcn_rcpf)
  return __builtin_amdgcn_rcpf(x);
#else
  return 1.0f / x;
#endif
}
__device__ __forceinline__ float fast_tanh(float x) {
#if __has_builtin(__builtin_amdgcn_tanhf)
  return __builtin_amdgcn_tanhf(x);
#else
  const float e = __expf(2.0f * x);
  return 1.0f - 2.0f * fast_rcp(e + 1.0f);
#endif
}
__device__ __forceinline__ float fast_sigmoid(float x) {
  return fast_rcp(1.0f + __expf(-x));
}

// Block = 128 threads (4 waves), batch tile of 32 (two 16-row WMMA tiles).
// Wave w owns hidden slice [16w,16w+16): 18 v_wmma_f32_16x16x32_f16 per step.
// All three gate GEMMs fully on WMMA; the n-gate keeps a SEPARATE accumulator
// for its input-side term (anx) since r multiplies only the hidden-side term:
//   n = tanh(anx + r * (anh + b_hh_n)).
__global__ __launch_bounds__(128)
void hybrid_selector_kernel(
    const float* __restrict__ H_norm,  const float* __restrict__ X_phys,
    const float* __restrict__ dt_val,  const float* __restrict__ gumbel,
    const float* __restrict__ W_ih,    const float* __restrict__ W_hh,
    const float* __restrict__ b_ih,    const float* __restrict__ b_hh,
    const float* __restrict__ W_head,  const float* __restrict__ b_head,
    const float* __restrict__ bank_W1, const float* __restrict__ bank_b1,
    const float* __restrict__ bank_W2, const float* __restrict__ bank_b2,
    float* __restrict__ out)
{
  __shared__ _Float16 sh_h[BT * HSTR];    // augmented state rows (f16)
  __shared__ float    sh_hf[BT * HID];    // final h (f32)
  __shared__ float    sh_logits[BT * MM];
  __shared__ int      sh_idx[BT];
  __shared__ float    sh_hid[BT * HB];    // bank layer-1 activations

  const int tid  = threadIdx.x;
  const int lane = tid & 31;
  const int wv   = tid >> 5;            // wave id 0..3 -> hidden slice
  const int b0   = blockIdx.x * BT;     // batch tile base
  const int lo16 = lane & 15;
  const int hi   = (lane >> 4) & 1;     // lane group (0: lanes 0-15)

  // Hidden column this lane owns in every C/D tile (C/D: lane = N column).
  const int j = 16 * wv + lo16;

  const float bhn = b_hh[j + 128];      // n hidden bias (applied inside r*)

  // ---- B-fragments, VGPR-resident for the whole T loop ----
  // 16-bit B 32x16 (KxN): lane -> column n = lane&15; lanes 0-15 hold K+0..15,
  // lanes 16-31 hold K+16..31 (contiguous K order chosen here; A uses the same).
  // bfh[g*2+c]: W_hh^T chunks (K=0..63) for gates g=r,z,n.
  // bfx[g]    : augmented chunk (K=64..95) = [W_ih^T | bias | 0].
  //             bias = b_ih+b_hh for r,z; b_ih only for n.
  v16h bfh[6], bfx[3];
  #pragma unroll
  for (int g = 0; g < 3; ++g) {
    const int col = 64 * g + j;                 // gate output column
    #pragma unroll
    for (int c = 0; c < 2; ++c) {
      const int kb = 32 * c + 16 * hi;
      v16h tf;
      #pragma unroll
      for (int q = 0; q < 16; ++q)
        tf[q] = (_Float16)W_hh[col * HID + kb + q];
      bfh[g * 2 + c] = tf;
    }
    {
      const float bias = (g < 2) ? (b_ih[col] + b_hh[col]) : b_ih[col];
      v16h tf;
      #pragma unroll
      for (int q = 0; q < 16; ++q) {
        const int K = 16 * hi + q;              // K-64 within augmented chunk
        const float val = (K < NN) ? W_ih[col * NN + K]
                                   : ((K == NN) ? bias : 0.0f);
        tf[q] = (_Float16)val;
      }
      bfx[g] = tf;
    }
  }

  // ---- init: augmented row = [h=0(64) | x | 1 | 0...] ----
  float hprev[16];                       // 8 rows x 2 batch tiles
  #pragma unroll
  for (int v = 0; v < 16; ++v) hprev[v] = 0.0f;
  for (int i = tid; i < BT * HSTR; i += 128) sh_h[i] = (_Float16)0.0f;
  __syncthreads();
  if (tid < BT) sh_h[tid * HSTR + HID + NN] = (_Float16)1.0f;  // constant-1 col

  // x double buffer: load row[t+1] while storing row[t].
  const float4* xrow = (const float4*)&H_norm[(size_t)(b0 + (tid & (BT - 1))) * (T_LEN * NN)];
  float4 xcur;
  if (tid < BT) xcur = xrow[0];

  // ================= GRU recurrence over T steps =================
  for (int t = 0; t < T_LEN; ++t) {
    if (tid < BT) {
      _Float16* xr = &sh_h[tid * HSTR + HID];
      xr[0] = (_Float16)xcur.x; xr[1] = (_Float16)xcur.y;
      xr[2] = (_Float16)xcur.z; xr[3] = (_Float16)xcur.w;
      if (t + 1 < T_LEN) xcur = xrow[t + 1];
      if (t + 2 < T_LEN) __builtin_prefetch(&xrow[t + 2], 0, 0);
    }
    __syncthreads();   // x(t) and h(t-1) visible

    // A fragments (16-bit A 16x32): lane = batch row m = lane&15 (+16 tile1);
    // lanes 0-15: K 0-15; lanes 16-31: K 16-31 (matches B striping).
    v16h a0, a1, a2, c0f, c1f, c2f;
    {
      const _Float16* hrow0 = &sh_h[lo16 * HSTR];
      const _Float16* hrow1 = &sh_h[(16 + lo16) * HSTR];
      const int base0 = hi ? 16 : 0;
      #pragma unroll
      for (int q = 0; q < 8; ++q) {
        a0[q]      = hrow0[base0 + q];
        a0[8 + q]  = hrow0[base0 + 8 + q];
        a1[q]      = hrow0[32 + base0 + q];
        a1[8 + q]  = hrow0[32 + base0 + 8 + q];
        a2[q]      = hrow0[64 + base0 + q];
        a2[8 + q]  = hrow0[64 + base0 + 8 + q];
        c0f[q]     = hrow1[base0 + q];
        c0f[8 + q] = hrow1[base0 + 8 + q];
        c1f[q]     = hrow1[32 + base0 + q];
        c1f[8 + q] = hrow1[32 + base0 + 8 + q];
        c2f[q]     = hrow1[64 + base0 + q];
        c2f[8 + q] = hrow1[64 + base0 + 8 + q];
      }
    }

    // Gate pre-activations, all on WMMA.
    // r/z: 3-chunk chain (h + x + bias). n: 2-chunk h-side + separate x-side.
    v8f ar0 = {}, az0 = {}, an0 = {}, ax0 = {};
    v8f ar1 = {}, az1 = {}, an1 = {}, ax1 = {};
    ar0 = __builtin_amdgcn_wmma_f32_16x16x32_f16(false, a0,  false, bfh[0], (short)0, ar0, false, false);
    ar1 = __builtin_amdgcn_wmma_f32_16x16x32_f16(false, c0f, false, bfh[0], (short)0, ar1, false, false);
    az0 = __builtin_amdgcn_wmma_f32_16x16x32_f16(false, a0,  false, bfh[2], (short)0, az0, false, false);
    az1 = __builtin_amdgcn_wmma_f32_16x16x32_f16(false, c0f, false, bfh[2], (short)0, az1, false, false);
    an0 = __builtin_amdgcn_wmma_f32_16x16x32_f16(false, a0,  false, bfh[4], (short)0, an0, false, false);
    an1 = __builtin_amdgcn_wmma_f32_16x16x32_f16(false, c0f, false, bfh[4], (short)0, an1, false, false);
    ax0 = __builtin_amdgcn_wmma_f32_16x16x32_f16(false, a2,  false, bfx[2], (short)0, ax0, false, false);
    ax1 = __builtin_amdgcn_wmma_f32_16x16x32_f16(false, c2f, false, bfx[2], (short)0, ax1, false, false);
    ar0 = __builtin_amdgcn_wmma_f32_16x16x32_f16(false, a1,  false, bfh[1], (short)0, ar0, false, false);
    ar1 = __builtin_amdgcn_wmma_f32_16x16x32_f16(false, c1f, false, bfh[1], (short)0, ar1, false, false);
    az0 = __builtin_amdgcn_wmma_f32_16x16x32_f16(false, a1,  false, bfh[3], (short)0, az0, false, false);
    az1 = __builtin_amdgcn_wmma_f32_16x16x32_f16(false, c1f, false, bfh[3], (short)0, az1, false, false);
    an0 = __builtin_amdgcn_wmma_f32_16x16x32_f16(false, a1,  false, bfh[5], (short)0, an0, false, false);
    an1 = __builtin_amdgcn_wmma_f32_16x16x32_f16(false, c1f, false, bfh[5], (short)0, an1, false, false);
    ar0 = __builtin_amdgcn_wmma_f32_16x16x32_f16(false, a2,  false, bfx[0], (short)0, ar0, false, false);
    ar1 = __builtin_amdgcn_wmma_f32_16x16x32_f16(false, c2f, false, bfx[0], (short)0, ar1, false, false);
    az0 = __builtin_amdgcn_wmma_f32_16x16x32_f16(false, a2,  false, bfx[1], (short)0, az0, false, false);
    az1 = __builtin_amdgcn_wmma_f32_16x16x32_f16(false, c2f, false, bfx[1], (short)0, az1, false, false);

    // Gates + state update (lane owns column j, rows m). Pure elementwise.
    float hnew[16];
    #pragma unroll
    for (int bt = 0; bt < 2; ++bt) {
      #pragma unroll
      for (int v = 0; v < 8; ++v) {
        const int vi = 8 * bt + v;
        const float pr  = bt ? ar1[v] : ar0[v];   // full r pre-activation
        const float pz  = bt ? az1[v] : az0[v];   // full z pre-activation
        const float hn  = bt ? an1[v] : an0[v];   // h-side n term
        const float gin = bt ? ax1[v] : ax0[v];   // x-side n term (+ b_ih_n)
        const float r = fast_sigmoid(pr);
        const float z = fast_sigmoid(pz);
        const float n = fast_tanh(gin + r * (hn + bhn));
        hnew[vi]  = (1.0f - z) * n + z * hprev[vi];
        hprev[vi] = hnew[vi];
      }
    }
    __syncthreads();   // everyone done reading sh_h
    #pragma unroll
    for (int bt = 0; bt < 2; ++bt) {
      #pragma unroll
      for (int v = 0; v < 8; ++v) {
        const int m = 16 * bt + v + (hi ? 8 : 0);
        sh_h[m * HSTR + j] = (_Float16)hnew[8 * bt + v];
      }
    }
  }

  // ---- stash final h in f32 for the head ----
  #pragma unroll
  for (int bt = 0; bt < 2; ++bt) {
    #pragma unroll
    for (int v = 0; v < 8; ++v) {
      const int m = 16 * bt + v + (hi ? 8 : 0);
      sh_hf[m * HID + j] = hprev[8 * bt + v];
    }
  }
  __syncthreads();

  const float dt = dt_val[0];

  // Output sections: [pred | logits | onehot | idx]
  float* out_pred   = out;
  float* out_logits = out + (size_t)B_TOT * 4;
  float* out_onehot = out + (size_t)B_TOT * 8;
  float* out_idx    = out + (size_t)B_TOT * 12;

  // ---- head: logits = h @ W_head^T + b_head  (128 threads = 32x4) ----
  {
    const int m = tid >> 2, mo = tid & 3;
    float acc = b_head[mo];
    #pragma unroll 8
    for (int k = 0; k < HID; ++k) acc += sh_hf[m * HID + k] * W_head[mo * HID + k];
    sh_logits[m * MM + mo] = acc;
    out_logits[(size_t)(b0 + m) * MM + mo] = acc;
  }
  __syncthreads();

  // ---- straight-through gumbel: idx = argmax(logits + g); onehot == y_hard ----
  if (tid < BT) {
    const int m = tid;
    int best = 0;
    float bv = sh_logits[m * MM] + gumbel[(size_t)(b0 + m) * MM];
    #pragma unroll
    for (int k = 1; k < MM; ++k) {
      const float v = sh_logits[m * MM + k] + gumbel[(size_t)(b0 + m) * MM + k];
      if (v > bv) { bv = v; best = k; }
    }
    sh_idx[m] = best;
    out_idx[b0 + m] = (float)best;
    #pragma unroll
    for (int k = 0; k < MM; ++k)
      out_onehot[(size_t)(b0 + m) * MM + k] = (k == best) ? 1.0f : 0.0f;
  }
  __syncthreads();

  // ---- bank layer 1: hid = tanh([x,dt] @ W1[e] + b1[e]) (per-sample expert) ----
  {
    const int s  = tid >> 2;     // sample 0..31
    const int ts = tid & 3;
    const int e  = sh_idx[s];
    const float x0 = X_phys[(size_t)(b0 + s) * NN + 0];
    const float x1 = X_phys[(size_t)(b0 + s) * NN + 1];
    const float x2 = X_phys[(size_t)(b0 + s) * NN + 2];
    const float x3 = X_phys[(size_t)(b0 + s) * NN + 3];
    const float* W1 = &bank_W1[(size_t)e * 5 * HB];
    const float* b1 = &bank_b1[(size_t)e * HB];
    #pragma unroll 4
    for (int c = 0; c < HB / 4; ++c) {
      const int hh = c * 4 + ts;
      const float acc = b1[hh]
          + x0 * W1[0 * HB + hh] + x1 * W1[1 * HB + hh]
          + x2 * W1[2 * HB + hh] + x3 * W1[3 * HB + hh]
          + dt * W1[4 * HB + hh];
      sh_hid[s * HB + hh] = fast_tanh(acc);
    }
  }
  __syncthreads();

  // ---- bank layer 2 + Euler step: pred = x + dt * (hid @ W2[e] + b2[e]) ----
  {
    const int s = tid >> 2, o = tid & 3;
    const int e = sh_idx[s];
    const float* W2 = &bank_W2[(size_t)e * HB * NN];
    float acc = bank_b2[e * NN + o];
    #pragma unroll 8
    for (int h = 0; h < HB; ++h) acc += sh_hid[s * HB + h] * W2[h * NN + o];
    const float xp = X_phys[(size_t)(b0 + s) * NN + o];
    out_pred[(size_t)(b0 + s) * NN + o] = xp + dt * acc;
  }
}

extern "C" void kernel_launch(void* const* d_in, const int* in_sizes, int n_in,
                              void* d_out, int out_size, void* d_ws, size_t ws_size,
                              hipStream_t stream) {
  (void)in_sizes; (void)n_in; (void)out_size; (void)d_ws; (void)ws_size;
  hybrid_selector_kernel<<<B_TOT / BT, 128, 0, stream>>>(
      (const float*)d_in[0],  (const float*)d_in[1],  (const float*)d_in[2],
      (const float*)d_in[3],  (const float*)d_in[4],  (const float*)d_in[5],
      (const float*)d_in[6],  (const float*)d_in[7],  (const float*)d_in[8],
      (const float*)d_in[9],  (const float*)d_in[10], (const float*)d_in[11],
      (const float*)d_in[12], (const float*)d_in[13], (float*)d_out);
}
